// ScannedRNN_26628797235775
// MI455X (gfx1250) — matrix operand order repaired and checked by reading, not compiled, not run on hardware
//
#include <hip/hip_runtime.h>
#include <stdint.h>

// GRU scan, MI455X (gfx1250, wave32, WMMA bf16).
//   Phase 0: transpose Wi/Wh -> N-major bf16 (B-fragment friendly).
//   Phase 1: GI = X @ Wi as one big parallel WMMA GEMM; sched_group_barrier
//            pins "all loads then all WMMAs" per k-step so the backend can't
//            wait-serialize on a single recycled B register set.
//   Phase 2: per-batch-group independent recurrence, 8 WGs x 32 waves,
//            h in LDS (bf16); fragments double-buffered across k-steps with
//            sched_group_barrier so loads run one iteration ahead of WMMAs.

typedef __attribute__((ext_vector_type(16))) __bf16 v16bf;
typedef __attribute__((ext_vector_type(8)))  float  v8f;
typedef __attribute__((ext_vector_type(4)))  float  f32x4;

#define T_  512
#define B_  128
#define H_  512
#define H3  1536
#define LDH 520   // LDS row stride in halves: 260 words % 64 banks = 4 -> conflict-free

// sched_group_barrier masks
#define SG_WMMA   0x008
#define SG_VMEMRD 0x020
#define SG_DSRD   0x100

static __device__ __forceinline__ unsigned short f2bf(float f) {
  union { float f; unsigned u; } c; c.f = f;
  unsigned u = c.u;
  return (unsigned short)((u + 0x7FFFu + ((u >> 16) & 1u)) >> 16);  // RNE
}
static __device__ __forceinline__ float bf2f(unsigned short h) {
  union { unsigned u; float f; } c; c.u = ((unsigned)h) << 16;
  return c.f;
}
static __device__ __forceinline__ float sigmoidf_(float x) {
  return 1.0f / (1.0f + __expf(-x));
}

union FragBF { v16bf v; uint4 q[2]; };

// ---------------------------------------------------------------------------
// Phase 0: Wi,Wh [H, 3H] f32 row-major -> WiT,WhT [3H][H] bf16 (N-major).
// ---------------------------------------------------------------------------
__global__ void prep_transpose(const float* __restrict__ Wi,
                               const float* __restrict__ Wh,
                               unsigned short* __restrict__ WiT,
                               unsigned short* __restrict__ WhT) {
  int e = blockIdx.x * blockDim.x + threadIdx.x;   // 0 .. 2*H*H3-1
  const int total = H_ * H3;
  const float* src = (e < total) ? Wi : Wh;
  unsigned short* dst = (e < total) ? WiT : WhT;
  int r = (e < total) ? e : e - total;
  int n = r / H_;
  int k = r - n * H_;
  dst[n * H_ + k] = f2bf(src[k * H3 + n]);
}

// ---------------------------------------------------------------------------
// Phase 1: GI[65536, 1536] = X[65536, 512] @ Wi.  512 thr (16 waves) /block,
// block owns 16 rows (A staged f32->bf16 in LDS); each wave does 6 N-tiles.
// ---------------------------------------------------------------------------
__global__ void __launch_bounds__(512) gi_gemm(const float* __restrict__ X,
                                               const unsigned short* __restrict__ WiT,
                                               float* __restrict__ GI) {
  __shared__ unsigned short As[16 * LDH];
  const int tid = threadIdx.x;
  const long grow0 = (long)blockIdx.x * 16;

  // Stage A tile: 16 x 512 f32 -> bf16 LDS (x is read exactly once -> NT)
  {
    int row  = tid >> 5;          // 32 threads per row
    int col0 = (tid & 31) * 16;
    const f32x4* src = (const f32x4*)(X + (grow0 + row) * H_ + col0);
    unsigned short* dst = As + row * LDH + col0;
#pragma unroll
    for (int j = 0; j < 4; ++j) {
      f32x4 v = __builtin_nontemporal_load(src + j);
      uint2 p;
      p.x = (unsigned)f2bf(v.x) | ((unsigned)f2bf(v.y) << 16);
      p.y = (unsigned)f2bf(v.z) | ((unsigned)f2bf(v.w) << 16);
      *(uint2*)(dst + j * 4) = p;
    }
  }
  __syncthreads();

  const int wave = tid >> 5, lane = tid & 31;       // 16 waves
  const int lane15 = lane & 15, hi = lane >> 4;
  // A layout (16-bit, 16x32): lanes 0-15 M=lane K=kb+0..7 & kb+16..23,
  //                           lanes 16-31 M=lane-16 K=kb+8..15 & kb+24..31.
  const unsigned short* arow = As + lane15 * LDH + hi * 8;

  // Precompute the 6 B column-base pointers (hoist the 64-bit address math).
  const unsigned short* bbase[6];
#pragma unroll
  for (int j = 0; j < 6; ++j)
    bbase[j] = WiT + (long)((wave * 6 + j) * 16 + lane15) * H_ + hi * 8;

  v8f acc[6];
#pragma unroll
  for (int j = 0; j < 6; ++j) acc[j] = {};

#pragma unroll 2
  for (int kb = 0; kb < H_; kb += 32) {
    FragBF a;
    a.q[0] = *(const uint4*)(arow + kb);
    a.q[1] = *(const uint4*)(arow + kb + 16);
    FragBF b[6];
#pragma unroll
    for (int j = 0; j < 6; ++j) {          // all 12 loads issue as one clause
      b[j].q[0] = *(const uint4*)(bbase[j] + kb);
      b[j].q[1] = *(const uint4*)(bbase[j] + kb + 16);
    }
#pragma unroll
    for (int j = 0; j < 6; ++j)            // waits drain incrementally
      acc[j] = __builtin_amdgcn_wmma_f32_16x16x32_bf16(
          false, a.v, false, b[j].v, (short)0, acc[j], false, false);
    // pin the shape: [2 ds reads][12 vmem reads][6 wmma] per k-step
    __builtin_amdgcn_sched_group_barrier(SG_DSRD,   2, 0);
    __builtin_amdgcn_sched_group_barrier(SG_VMEMRD, 12, 0);
    __builtin_amdgcn_sched_group_barrier(SG_WMMA,   6, 0);
  }
  // C layout: VGPR i -> M = i + 8*hi, N = lane15
#pragma unroll
  for (int j = 0; j < 6; ++j) {
    int n0 = (wave * 6 + j) * 16;
#pragma unroll
    for (int i = 0; i < 8; ++i)
      GI[(grow0 + i + hi * 8) * (long)H3 + n0 + lane15] = acc[j][i];
  }
}

// ---------------------------------------------------------------------------
// Phase 2: recurrence. 8 blocks x 1024 threads; block owns batch rows
// [16*blk, 16*blk+16). h lives in LDS as bf16. Wave w owns columns
// c=16w..16w+15 and computes gate tiles r(c), z(512+c), n(1024+c).
// Fragments are double-buffered across the 16 k-steps.
// ---------------------------------------------------------------------------
__global__ void __launch_bounds__(1024) gru_scan(const float* __restrict__ GI,
                                                 const unsigned short* __restrict__ WhT,
                                                 const unsigned char* __restrict__ resets,
                                                 const float* __restrict__ h0,
                                                 const float* __restrict__ bh,
                                                 float* __restrict__ out) {
  __shared__ unsigned short hs[16 * LDH];
  __shared__ unsigned char  rs[T_ * 16];      // this block's resets column
  const int tid = threadIdx.x;
  const int b0  = blockIdx.x * 16;

  const int zrow = tid >> 6;          // 16 rows, 64 threads per row
  const int zcol = (tid & 63) * 8;    // 8 halves (16 B) per thread

  // init h <- h0 (f32 -> bf16)
  {
    const f32x4* src = (const f32x4*)(h0 + (long)(b0 + zrow) * H_ + zcol);
    f32x4 v0 = src[0], v1 = src[1];
    uint4 p;
    p.x = (unsigned)f2bf(v0.x) | ((unsigned)f2bf(v0.y) << 16);
    p.y = (unsigned)f2bf(v0.z) | ((unsigned)f2bf(v0.w) << 16);
    p.z = (unsigned)f2bf(v1.x) | ((unsigned)f2bf(v1.y) << 16);
    p.w = (unsigned)f2bf(v1.z) | ((unsigned)f2bf(v1.w) << 16);
    *(uint4*)(hs + zrow * LDH + zcol) = p;
  }
  // stage resets[t][b0..b0+15] -> LDS once (off the serial critical path)
#pragma unroll
  for (int i = 0; i < (T_ * 16) / 1024; ++i) {
    int idx = i * 1024 + tid;
    rs[idx] = resets[(idx >> 4) * B_ + b0 + (idx & 15)];
  }

  const int wave = tid >> 5, lane = tid & 31;
  const int lane15 = lane & 15, hi = lane >> 4;
  const int c = wave * 16;

  const unsigned short* arow = hs + lane15 * LDH + hi * 8;
  const unsigned short* bpr = WhT + (long)(c + lane15) * H_ + hi * 8;
  const unsigned short* bpz = WhT + (long)(512 + c + lane15) * H_ + hi * 8;
  const unsigned short* bpn = WhT + (long)(1024 + c + lane15) * H_ + hi * 8;
  const float bhr = bh[c + lane15];
  const float bhz = bh[512 + c + lane15];
  const float bhn = bh[1024 + c + lane15];

  __syncthreads();

  for (int t = 0; t < T_; ++t) {
    // prefetch this step's GI tile rows early: these global loads fly while
    // the reset barrier and the whole 16-step WMMA loop execute.
    float gir[8], giz[8], gin[8];
#pragma unroll
    for (int i = 0; i < 8; ++i) {
      long gib = ((long)t * B_ + b0 + i + hi * 8) * H3 + lane15;
      gir[i] = GI[gib + c];
      giz[i] = GI[gib + 512 + c];
      gin[i] = GI[gib + 1024 + c];
    }

    // episode reset: zero this block's h rows where resets[t][b]
    if (rs[t * 16 + zrow]) {
      uint4 z = {0u, 0u, 0u, 0u};
      *(uint4*)(hs + zrow * LDH + zcol) = z;
    }
    __syncthreads();

    v8f ar = {}, az = {}, an = {};
    {
      FragBF ab[2], rb[2], zb[2], nb[2];
      // preload k-step 0
      ab[0].q[0] = *(const uint4*)(arow);       ab[0].q[1] = *(const uint4*)(arow + 16);
      rb[0].q[0] = *(const uint4*)(bpr);        rb[0].q[1] = *(const uint4*)(bpr + 16);
      zb[0].q[0] = *(const uint4*)(bpz);        zb[0].q[1] = *(const uint4*)(bpz + 16);
      nb[0].q[0] = *(const uint4*)(bpn);        nb[0].q[1] = *(const uint4*)(bpn + 16);
#pragma unroll
      for (int j = 0; j < 16; ++j) {            // kb = 32*j, fully unrolled
        const int s = j & 1, d = s ^ 1;
        if (j + 1 < 16) {                       // prefetch next k-step
          const int kn = (j + 1) * 32;
          ab[d].q[0] = *(const uint4*)(arow + kn);  ab[d].q[1] = *(const uint4*)(arow + kn + 16);
          rb[d].q[0] = *(const uint4*)(bpr + kn);   rb[d].q[1] = *(const uint4*)(bpr + kn + 16);
          zb[d].q[0] = *(const uint4*)(bpz + kn);   zb[d].q[1] = *(const uint4*)(bpz + kn + 16);
          nb[d].q[0] = *(const uint4*)(bpn + kn);   nb[d].q[1] = *(const uint4*)(bpn + kn + 16);
        }
        ar = __builtin_amdgcn_wmma_f32_16x16x32_bf16(false, ab[s].v, false, rb[s].v, (short)0, ar, false, false);
        az = __builtin_amdgcn_wmma_f32_16x16x32_bf16(false, ab[s].v, false, zb[s].v, (short)0, az, false, false);
        an = __builtin_amdgcn_wmma_f32_16x16x32_bf16(false, ab[s].v, false, nb[s].v, (short)0, an, false, false);
        // pin: [2 ds reads (next A)][6 vmem reads (next B)][3 wmma (current)]
        __builtin_amdgcn_sched_group_barrier(SG_DSRD,   2, 0);
        __builtin_amdgcn_sched_group_barrier(SG_VMEMRD, 6, 0);
        __builtin_amdgcn_sched_group_barrier(SG_WMMA,   3, 0);
      }
    }

    // read old h for this tile before anyone overwrites LDS
    float hold[8];
#pragma unroll
    for (int i = 0; i < 8; ++i)
      hold[i] = bf2f(hs[(i + hi * 8) * LDH + c + lane15]);
    __syncthreads();

    // gates + state update; write ys (streaming -> NT) and new h
#pragma unroll
    for (int i = 0; i < 8; ++i) {
      int m = i + hi * 8;
      float r  = sigmoidf_(gir[i] + ar[i] + bhr);
      float z  = sigmoidf_(giz[i] + az[i] + bhz);
      float nn = tanhf(gin[i] + r * (an[i] + bhn));
      float hnew = (1.0f - z) * nn + z * hold[i];
      hs[m * LDH + c + lane15] = f2bf(hnew);
      __builtin_nontemporal_store(hnew,
          out + ((long)t * B_ + b0 + m) * H_ + c + lane15);
    }
    __syncthreads();
  }
}

// ---------------------------------------------------------------------------
extern "C" void kernel_launch(void* const* d_in, const int* in_sizes, int n_in,
                              void* d_out, int out_size, void* d_ws, size_t ws_size,
                              hipStream_t stream) {
  const float*         x      = (const float*)d_in[0];
  const unsigned char* resets = (const unsigned char*)d_in[1];  // jnp bool -> 1 byte
  const float*         h0     = (const float*)d_in[2];
  const float*         Wi     = (const float*)d_in[3];
  const float*         Wh     = (const float*)d_in[4];
  const float*         bh     = (const float*)d_in[5];
  float*               out    = (float*)d_out;

  char* ws = (char*)d_ws;
  float*          GI  = (float*)ws;                                   // 512*128*1536*4 B
  unsigned short* WiT = (unsigned short*)(ws + (size_t)T_ * B_ * H3 * 4);
  unsigned short* WhT = WiT + (size_t)H3 * H_;

  prep_transpose<<<(2 * H_ * H3) / 256, 256, 0, stream>>>(Wi, Wh, WiT, WhT);
  gi_gemm<<<(T_ * B_) / 16, 512, 0, stream>>>(x, WiT, GI);
  gru_scan<<<B_ / 16, 1024, 0, stream>>>(GI, WhT, resets, h0, bh, out);
}
